// ScalarDistanceDeepSet_88622355186244
// MI455X (gfx1250) — matrix operand order repaired and checked by reading, not compiled, not run on hardware
//
#include <hip/hip_runtime.h>

typedef _Float16 half16 __attribute__((ext_vector_type(16)));
typedef float    float8 __attribute__((ext_vector_type(8)));

#define TOTAL_ROWS 1536000
#define NSEG 256
#define H2 128
#define NTILES (TOTAL_ROWS / 16)

// ---------------------------------------------------------------- zero scratch
__global__ void zero_kernel(float* __restrict__ p, int n) {
  int i = blockIdx.x * blockDim.x + threadIdx.x;
  if (i < n) p[i] = 0.f;
}

// ---------------------------------------------------------- fused phi + pooling
// h1 = relu(d*W1+b1) [16x64 f16, built in regs per WMMA A layout]
// h2 = relu(h1@W2+b2) via v_wmma_f32_16x16x32_f16, f32 accum
// pooled[seg] += h2 rows, register accumulator flushed on segment change.
__global__ void __launch_bounds__(256) phi_pool_kernel(
    const float* __restrict__ dist,   // [TOTAL,1]
    const int*   __restrict__ sids,   // [TOTAL] sorted
    const float* __restrict__ W1,     // [1,64]
    const float* __restrict__ b1,     // [64]
    const float* __restrict__ W2,     // [64,128]
    const float* __restrict__ b2,     // [128]
    float*       __restrict__ pooled) // [256,128]
{
  const int lane   = threadIdx.x & 31;
  const int laneLo = lane & 15;
  const int hi     = lane >> 4;   // 0: K 0-15 (A), rows 0-7 (C); 1: other half

  const int wid    = (blockIdx.x * blockDim.x + threadIdx.x) >> 5;
  const int nwaves = (gridDim.x * blockDim.x) >> 5;
  const int tpw    = (NTILES + nwaves - 1) / nwaves;
  int t0 = wid * tpw;
  int t1 = t0 + tpw; if (t1 > NTILES) t1 = NTILES;

  // Per-lane phi-layer-1 weights in A-fragment order:
  // K(e,kc) = kc*32 + hi*16 + e
  float w1v[32], b1v[32];
  #pragma unroll
  for (int kc = 0; kc < 2; ++kc)
    #pragma unroll
    for (int e = 0; e < 16; ++e) {
      int K = kc * 32 + hi * 16 + e;
      w1v[kc * 16 + e] = W1[K];
      b1v[kc * 16 + e] = b1[K];
    }

  // W2 as register-resident f16 B fragments (8 N-tiles x 2 K-chunks).
  // B order chosen here, contiguous k (32x16 per frag): lane<16 holds elem e = B[K=e][N=lane];
  // lane>=16 holds elem e = B[K=16+e][N=lane-16].
  half16 bfrag[8][2];
  #pragma unroll
  for (int nt = 0; nt < 8; ++nt)
    #pragma unroll
    for (int kc = 0; kc < 2; ++kc) {
      const int col = nt * 16 + laneLo;
      const int kb  = kc * 32 + hi * 16;
      #pragma unroll
      for (int e = 0; e < 16; ++e)
        bfrag[nt][kc][e] = (_Float16)W2[(kb + e) * H2 + col];
    }

  float b2v[8];
  #pragma unroll
  for (int nt = 0; nt < 8; ++nt) b2v[nt] = b2[nt * 16 + laneLo];

  // Register pooling accumulator for the current segment.
  float pool[8];
  #pragma unroll
  for (int nt = 0; nt < 8; ++nt) pool[nt] = 0.f;
  int cur_seg = -1;

  auto flush = [&]() {
    if (cur_seg >= 0) {
      #pragma unroll
      for (int nt = 0; nt < 8; ++nt) {
        atomicAdd(&pooled[cur_seg * H2 + nt * 16 + laneLo], pool[nt]);
        pool[nt] = 0.f;
      }
    }
  };

  for (int t = t0; t < t1; ++t) {
    const int base = t * 16;
    const float d  = dist[base + laneLo];
    const int s0   = sids[base];        // wave-uniform
    const int s15  = sids[base + 15];   // wave-uniform (sorted -> uniform iff equal)

    float8 acc[8] = {};

    #pragma unroll
    for (int kc = 0; kc < 2; ++kc) {
      half16 af;
      #pragma unroll
      for (int e = 0; e < 16; ++e) {
        float hv = fmaf(d, w1v[kc * 16 + e], b1v[kc * 16 + e]);
        hv = hv > 0.f ? hv : 0.f;
        af[e] = (_Float16)hv;
      }
      #pragma unroll
      for (int nt = 0; nt < 8; ++nt)
        acc[nt] = __builtin_amdgcn_wmma_f32_16x16x32_f16(
            false, af, false, bfrag[nt][kc], (short)0, acc[nt], false, false);
    }

    if (s0 == s15) {                    // whole tile in one segment (common)
      if (s0 != cur_seg) { flush(); cur_seg = s0; }
      #pragma unroll
      for (int nt = 0; nt < 8; ++nt) {
        float v = 0.f;
        #pragma unroll
        for (int r = 0; r < 8; ++r) {   // C layout: VGPR r -> row hi*8+r, col laneLo
          float hv = acc[nt][r] + b2v[nt];
          v += hv > 0.f ? hv : 0.f;
        }
        pool[nt] += v;
      }
    } else {                            // segment boundary tile (~255 total)
      int srow[8];
      #pragma unroll
      for (int r = 0; r < 8; ++r) srow[r] = sids[base + hi * 8 + r];
      #pragma unroll
      for (int nt = 0; nt < 8; ++nt)
        #pragma unroll
        for (int r = 0; r < 8; ++r) {
          float hv = acc[nt][r] + b2v[nt];
          hv = hv > 0.f ? hv : 0.f;
          atomicAdd(&pooled[srow[r] * H2 + nt * 16 + laneLo], hv);
        }
    }
  }
  flush();
}

// ----------------------------------------------------------------- rho layers
// Tiny dense layers (<= 17 MFLOP each): thread-per-output-element is fine.
__global__ void rho_kernel(const float* __restrict__ X, const float* __restrict__ W,
                           const float* __restrict__ bv, float* __restrict__ Y,
                           int M, int K, int N, int do_relu) {
  int gid = blockIdx.x * blockDim.x + threadIdx.x;
  if (gid >= M * N) return;
  int i = gid / N, j = gid - i * N;
  float s = bv[j];
  const float* x = X + i * K;
  #pragma unroll 8
  for (int k = 0; k < K; ++k) s = fmaf(x[k], W[k * N + j], s);
  if (do_relu) s = s > 0.f ? s : 0.f;
  Y[gid] = s;
}

// ------------------------------------------------------------------- launcher
extern "C" void kernel_launch(void* const* d_in, const int* in_sizes, int n_in,
                              void* d_out, int out_size, void* d_ws, size_t ws_size,
                              hipStream_t stream) {
  const float* dist = (const float*)d_in[0];
  const int*   sids = (const int*)  d_in[1];
  // d_in[2] = num_segments (hardcoded 256)
  const float* W1 = (const float*)d_in[3];
  const float* b1 = (const float*)d_in[4];
  const float* W2 = (const float*)d_in[5];
  const float* b2 = (const float*)d_in[6];
  const float* W3 = (const float*)d_in[7];
  const float* b3 = (const float*)d_in[8];
  const float* W4 = (const float*)d_in[9];
  const float* b4 = (const float*)d_in[10];
  const float* W5 = (const float*)d_in[11];
  const float* b5 = (const float*)d_in[12];

  float* pooled = (float*)d_ws;          // 256*128
  float* r1     = pooled + 256 * 128;    // 256*256
  float* r2     = r1     + 256 * 256;    // 256*128
  float* out    = (float*)d_out;         // 256*64

  zero_kernel<<<(256 * 128 + 255) / 256, 256, 0, stream>>>(pooled, 256 * 128);

  // 240 blocks * 8 waves = 1920 waves; 96000 tiles / 1920 = 50 tiles per wave.
  phi_pool_kernel<<<240, 256, 0, stream>>>(dist, sids, W1, b1, W2, b2, pooled);

  rho_kernel<<<(256 * 256 + 255) / 256, 256, 0, stream>>>(pooled, W3, b3, r1, 256, 128, 256, 1);
  rho_kernel<<<(256 * 128 + 255) / 256, 256, 0, stream>>>(r1,     W4, b4, r2, 256, 256, 128, 1);
  rho_kernel<<<(256 * 64  + 255) / 256, 256, 0, stream>>>(r2,     W5, b5, out, 256, 128, 64, 0);
}
